// NeuralMeshSimplification_72911364817261
// MI455X (gfx1250) — compile-verified, matrix operand order checked by reading, not run on hardware
//
#include <hip/hip_runtime.h>
#include <hip/hip_bf16.h>
#include <stdint.h>

typedef __attribute__((ext_vector_type(16))) _Float16 v16h;
typedef __attribute__((ext_vector_type(8)))  float    v8f;

namespace {
constexpr int Nn = 20000;
constexpr int Ee = 320000;
constexpr int Dd = 3;
constexpr int Hh = 128;
constexpr int Zz = 64;
constexpr int Kk = 20;
constexpr int Mm = 10000;     // N/2
constexpr int PAIRS = 190;    // K*(K-1)/2
constexpr int KTILES = Hh / 32;          // 4 wmma steps along K
constexpr int NTILES = Zz / 16;          // 4
constexpr int MTILES = Mm / 16;          // 625
constexpr int TILES  = MTILES * NTILES;  // 2500
}

__device__ __forceinline__ float sig(float x){ return 1.0f/(1.0f + __expf(-x)); }

__global__ void zero_kernel(float* p, long n){
  long i = (long)blockIdx.x*blockDim.x + threadIdx.x;
  long stride = (long)gridDim.x*blockDim.x;
  for (; i < n; i += stride) p[i] = 0.0f;
}

__global__ void segsum_kernel(const float* __restrict__ x, const int* __restrict__ ei,
                              float* __restrict__ agg){
  int e = blockIdx.x*blockDim.x + threadIdx.x;
  if (e >= Ee) return;
  int s = ei[e], d = ei[Ee + e];
  #pragma unroll
  for (int c = 0; c < Dd; ++c) atomicAdd(&agg[d*Dd + c], x[s*Dd + c]);
}

__global__ void hidden_kernel(const float* __restrict__ x, const float* __restrict__ agg,
                              const float* __restrict__ Wp1, const float* __restrict__ bp1,
                              float* __restrict__ h){
  long t = (long)blockIdx.x*blockDim.x + threadIdx.x;
  if (t >= (long)Nn*Hh) return;
  int n = (int)(t / Hh), j = (int)(t % Hh);
  float s = bp1[j];
  #pragma unroll
  for (int c = 0; c < Dd; ++c) s += (x[n*Dd + c] + agg[n*Dd + c]) * Wp1[c*Hh + j];
  h[t] = fmaxf(s, 0.0f);
}

__global__ void probs_kernel(const float* __restrict__ h, const float* __restrict__ wp2,
                             float* __restrict__ probs, float* __restrict__ out_sp){
  int n = blockIdx.x*blockDim.x + threadIdx.x;
  if (n >= Nn) return;
  const float* hr = h + (long)n*Hh;
  float s = 0.0f;
  #pragma unroll 8
  for (int j = 0; j < Hh; ++j) s += hr[j]*wp2[j];
  float p = sig(s);
  probs[n] = p; out_sp[n] = p;
}

// Exact sorted top-M by rank counting: probs (80KB) is L2-resident so the O(N^2)
// scan is cheap. Matches jax top_k tie semantics (descending, lower index first).
__global__ void rank_kernel(const float* __restrict__ probs, int* __restrict__ idxbuf,
                            float* __restrict__ out_idx){
  int n = blockIdx.x*blockDim.x + threadIdx.x;
  if (n >= Nn) return;
  float p = probs[n];
  int rank = 0;
  for (int j = 0; j < Nn; ++j){
    float q = probs[j];
    rank += (q > p) || (q == p && j < n);
  }
  if (rank < Mm){ idxbuf[rank] = n; out_idx[rank] = (float)n; }
}

__global__ void invinit_kernel(int* inv){
  int n = blockIdx.x*blockDim.x + threadIdx.x;
  if (n < Nn) inv[n] = -1;
}
__global__ void invset_kernel(const int* __restrict__ idxbuf, int* __restrict__ inv){
  int r = blockIdx.x*blockDim.x + threadIdx.x;
  if (r < Mm) inv[idxbuf[r]] = r;
}

__global__ void h1_kernel(const float* __restrict__ x, const int* __restrict__ idxbuf,
                          const float* __restrict__ We1, _Float16* __restrict__ h1){
  long t = (long)blockIdx.x*blockDim.x + threadIdx.x;
  if (t >= (long)Mm*Hh) return;
  int r = (int)(t / Hh), j = (int)(t % Hh);
  int node = idxbuf[r];
  float s = 0.0f;
  #pragma unroll
  for (int c = 0; c < Dd; ++c) s += x[node*Dd + c] * We1[c*Hh + j];
  h1[t] = (_Float16)fmaxf(s, 0.0f);
}

// Pre-pack We2 (128x64 f32) into the WMMA B per-lane layout as f16 pairs so the
// GEMM does only contiguous dword loads.
__global__ void packB_kernel(const float* __restrict__ We2, uint32_t* __restrict__ Bpack){
  int t = blockIdx.x*blockDim.x + threadIdx.x;
  if (t >= NTILES*KTILES*32*8) return;
  int j = t & 7, lane = (t >> 3) & 31, kb = (t >> 8) & 3, tn = t >> 10;
  int n = tn*16 + (lane & 15), half = lane >> 4;
  int kbase = (j < 4) ? 2*j : 16 + 2*(j - 4);
  int k = kb*32 + half*8 + kbase;
  union { uint32_t u; _Float16 h[2]; } cv;
  cv.h[0] = (_Float16)We2[(size_t)k*Zz + n];
  cv.h[1] = (_Float16)We2[(size_t)(k + 1)*Zz + n];
  Bpack[t] = cv.u;
}

// z[M x 64] = h1[M x 128] @ We2[128 x 64], f16 inputs, f32 accumulation.
// One wave per 16x16 output tile; K-loop of 4 x v_wmma_f32_16x16x32_f16.
__global__ void gemm_wmma_kernel(const _Float16* __restrict__ A16,
                                 const uint32_t* __restrict__ Bpack,
                                 float* __restrict__ z){
  int wave = threadIdx.x >> 5;
  int tile = blockIdx.x*(blockDim.x >> 5) + wave;
  if (tile >= TILES) return;             // uniform per wave -> EXEC stays all-1s
  int lane = threadIdx.x & 31;
  int tm = tile >> 2, tn = tile & 3;
  int mloc = lane & 15, half = lane >> 4;

  v8f c = {};
  // A layout (16-bit 16x32): lane = M (0..15), half selects K offset +8;
  // VGPR j<4 -> K=2j.., j>=4 -> K=16+2(j-4).. (all relative to half*8).
  const uint32_t* arow = (const uint32_t*)(A16 + (size_t)(tm*16 + mloc)*Hh + half*8);
  const uint32_t* bp   = Bpack + ((size_t)(tn*KTILES)*32 + lane)*8;

  #pragma unroll
  for (int kb = 0; kb < KTILES; ++kb){
    union { v16h v; uint32_t u[8]; } A, B;
    #pragma unroll
    for (int j = 0; j < 4; ++j){ A.u[j] = arow[j]; A.u[4 + j] = arow[8 + j]; }
    #pragma unroll
    for (int j = 0; j < 8; ++j){ B.u[j] = bp[j]; }
    if (kb + 1 < KTILES){
      __builtin_prefetch(arow + 16, 0, 1);   // next K-block of A row
      __builtin_prefetch(bp + 256, 0, 1);    // next packed B block
    }
    c = __builtin_amdgcn_wmma_f32_16x16x32_f16(
        /*neg_a=*/false, A.v, /*neg_b=*/false, B.v,
        /*c_mod=*/(short)0, c, /*reuse_a=*/false, /*reuse_b=*/false);
    arow += 16;    // +32 halves along K
    bp   += 256;   // next kb block in packed B
  }
  // C/D layout: VGPR r -> row M = r + 8*half, column = lane&15
  #pragma unroll
  for (int r = 0; r < 8; ++r)
    z[(size_t)(tm*16 + r + half*8)*Zz + tn*16 + mloc] = c[r];
}

__global__ void fprob_kernel(const float* __restrict__ x, const float* __restrict__ pos,
                             const int* __restrict__ idxbuf, const float* __restrict__ wfx,
                             const float* __restrict__ wfp, float* __restrict__ f){
  int r = blockIdx.x*blockDim.x + threadIdx.x;
  if (r >= Mm) return;
  int node = idxbuf[r];
  float s = 0.0f;
  #pragma unroll
  for (int c = 0; c < Dd; ++c)
    s += x[node*Dd + c]*wfx[c] + pos[node*Dd + c]*wfp[c];
  f[r] = sig(s);
}

__global__ void eprob_kernel(const int* __restrict__ ei, const int* __restrict__ inv,
                             const float* __restrict__ z, float* __restrict__ adj,
                             float* __restrict__ out_ep){
  int e = blockIdx.x*blockDim.x + threadIdx.x;
  if (e >= Ee) return;
  int iu = inv[ei[e]], iv = inv[ei[Ee + e]];
  bool valid = (iu >= 0) && (iv >= 0);
  int nu = valid ? iu : 0, nv = valid ? iv : 0;
  const float* zu = z + (size_t)nu*Zz;
  const float* zv = z + (size_t)nv*Zz;
  float s = 0.0f;
  #pragma unroll 8
  for (int c = 0; c < Zz; ++c) s += zu[c]*zv[c];
  float ep = sig(s);
  out_ep[e] = ep;
  // all adj values >= 0 so uint-bit atomicMax == float max
  if (valid && ep > 0.0f)
    atomicMax((unsigned int*)&adj[(size_t)nu*Mm + nv], __float_as_uint(ep));
}

// Row-wise exact top-20 with one LDS-resident copy of the 40KB row (320KB WGP LDS).
__global__ void knn_kernel(const float* __restrict__ adj, int* __restrict__ knn){
  __shared__ float srow[Mm];
  __shared__ float rv[256];
  __shared__ int   ri[256];
  int m = blockIdx.x;
  int tid = threadIdx.x;
  const float* row = adj + (size_t)m*Mm;
  for (int i = tid; i < Mm; i += blockDim.x) srow[i] = row[i];
  __syncthreads();
  for (int t = 0; t < Kk; ++t){
    float bv = -1.0f; int bi = Mm - 1;
    for (int i = tid; i < Mm; i += blockDim.x){
      float v = srow[i];
      if (v > bv || (v == bv && i < bi)){ bv = v; bi = i; }
    }
    rv[tid] = bv; ri[tid] = bi;
    __syncthreads();
    for (int s = blockDim.x >> 1; s > 0; s >>= 1){
      if (tid < s){
        float ov = rv[tid + s]; int oi = ri[tid + s];
        if (ov > rv[tid] || (ov == rv[tid] && oi < ri[tid])){ rv[tid] = ov; ri[tid] = oi; }
      }
      __syncthreads();
    }
    if (tid == 0){ knn[(size_t)m*Kk + t] = ri[0]; srow[ri[0]] = -2.0f; }
    __syncthreads();
  }
}

__global__ void tri_kernel(const float* __restrict__ adj, const int* __restrict__ knn,
                           const float* __restrict__ f, float* __restrict__ out_tri,
                           float* __restrict__ out_tp, float* __restrict__ out_fp,
                           float* __restrict__ out_mask){
  __shared__ int kn[Kk];
  __shared__ float fm_s;
  int m = blockIdx.x;
  int t = threadIdx.x;
  if (t < Kk) kn[t] = knn[(size_t)m*Kk + t];
  if (t == 0) fm_s = f[m];
  __syncthreads();
  if (t >= PAIRS) return;
  // triu_indices(K,1) row-major decode
  int p = t, jj = 0, span = Kk - 1;
  while (p >= span){ p -= span; ++jj; --span; }
  int ll = jj + 1 + p;
  int n1 = kn[jj], n2 = kn[ll];
  float a1  = adj[(size_t)m*Mm + n1];
  float a2  = adj[(size_t)m*Mm + n2];
  float a12 = adj[(size_t)n1*Mm + n2];
  bool valid = a12 > 0.0f;
  float tp = valid ? cbrtf(fmaxf(a1*a2*a12, 1e-12f)) : 0.0f;
  float fp = (fm_s + f[n1] + f[n2]) * (1.0f/3.0f) * (valid ? 1.0f : 0.0f);
  long base = (long)m*PAIRS + t;
  out_tri[base*3 + 0] = (float)m;
  out_tri[base*3 + 1] = (float)n1;
  out_tri[base*3 + 2] = (float)n2;
  out_tp[base]   = tp;
  out_fp[base]   = fp;
  out_mask[base] = (fp > 0.5f && valid) ? 1.0f : 0.0f;
}

extern "C" void kernel_launch(void* const* d_in, const int* in_sizes, int n_in,
                              void* d_out, int out_size, void* d_ws, size_t ws_size,
                              hipStream_t stream){
  (void)in_sizes; (void)n_in; (void)out_size; (void)ws_size;
  const float* x   = (const float*)d_in[0];
  const float* pos = (const float*)d_in[1];
  const int*   ei  = (const int*)d_in[2];
  const float* Wp1 = (const float*)d_in[3];
  const float* bp1 = (const float*)d_in[4];
  const float* wp2 = (const float*)d_in[5];
  const float* We1 = (const float*)d_in[6];
  const float* We2 = (const float*)d_in[7];
  const float* wfx = (const float*)d_in[8];
  const float* wfp = (const float*)d_in[9];

  float* out      = (float*)d_out;
  float* out_sp   = out;                                   // N
  float* out_idx  = out_sp + Nn;                           // M
  float* out_ep   = out_idx + Mm;                          // E
  float* out_tri  = out_ep + Ee;                           // M*190*3
  float* out_tp   = out_tri + (size_t)Mm*PAIRS*3;          // M*190
  float* out_fp   = out_tp + (size_t)Mm*PAIRS;             // M*190
  float* out_mask = out_fp + (size_t)Mm*PAIRS;             // M*190

  char* w = (char*)d_ws; size_t off = 0;
  auto alloc = [&](size_t bytes)->void*{
    void* p = w + off; off += (bytes + 255) & ~(size_t)255; return p;
  };
  float*    agg   = (float*)alloc((size_t)Nn*Dd*4);
  float*    h     = (float*)alloc((size_t)Nn*Hh*4);
  float*    probs = (float*)alloc((size_t)Nn*4);
  int*      idxb  = (int*)alloc((size_t)Mm*4);
  int*      inv   = (int*)alloc((size_t)Nn*4);
  _Float16* h1    = (_Float16*)alloc((size_t)Mm*Hh*2);
  uint32_t* Bpack = (uint32_t*)alloc((size_t)NTILES*KTILES*32*8*4);
  float*    z     = (float*)alloc((size_t)Mm*Zz*4);
  float*    fpb   = (float*)alloc((size_t)Mm*4);
  int*      knn   = (int*)alloc((size_t)Mm*Kk*4);
  float*    adj   = (float*)alloc((size_t)Mm*Mm*4);        // 400 MB

  const int B = 256;
  zero_kernel<<<256, B, 0, stream>>>(agg, (long)Nn*Dd);
  segsum_kernel<<<(Ee + B - 1)/B, B, 0, stream>>>(x, ei, agg);
  hidden_kernel<<<(unsigned)(((long)Nn*Hh + B - 1)/B), B, 0, stream>>>(x, agg, Wp1, bp1, h);
  probs_kernel<<<(Nn + B - 1)/B, B, 0, stream>>>(h, wp2, probs, out_sp);
  rank_kernel<<<(Nn + B - 1)/B, B, 0, stream>>>(probs, idxb, out_idx);
  invinit_kernel<<<(Nn + B - 1)/B, B, 0, stream>>>(inv);
  invset_kernel<<<(Mm + B - 1)/B, B, 0, stream>>>(idxb, inv);
  h1_kernel<<<(unsigned)(((long)Mm*Hh + B - 1)/B), B, 0, stream>>>(x, idxb, We1, h1);
  packB_kernel<<<(NTILES*KTILES*32*8 + B - 1)/B, B, 0, stream>>>(We2, Bpack);
  gemm_wmma_kernel<<<(TILES + 7)/8, 256, 0, stream>>>(h1, Bpack, z);
  fprob_kernel<<<(Mm + B - 1)/B, B, 0, stream>>>(x, pos, idxb, wfx, wfp, fpb);
  zero_kernel<<<4096, B, 0, stream>>>(adj, (long)Mm*Mm);
  eprob_kernel<<<(Ee + B - 1)/B, B, 0, stream>>>(ei, inv, z, adj, out_ep);
  knn_kernel<<<Mm, 256, 0, stream>>>(adj, knn);
  tri_kernel<<<Mm, 256, 0, stream>>>(adj, knn, fpb, out_tri, out_tp, out_fp, out_mask);
}
